// BinaryDiffRow_76089640615939
// MI455X (gfx1250) — compile-verified, matrix operand order checked by reading, not run on hardware
//
#include <hip/hip_runtime.h>
#include <stdint.h>

// ---------------------------------------------------------------------------
// BinaryDiffRow on MI455X (gfx1250):
//   y = x @ (base_t + coeff[:,None] * signs(mask))
// One fused GEMM, M=8192 K=4096 N=4096, split-precision bf16x3
// (xh*wh + xh*wl + xl*wh, fp32 accum) on v_wmma_f32_16x16x32_bf16,
// double-buffered LDS with async global->LDS staging (ASYNCcnt-tracked).
// ---------------------------------------------------------------------------

typedef __attribute__((ext_vector_type(16))) __bf16 v16bf;
typedef __attribute__((ext_vector_type(8)))  __bf16 v8bf;
typedef __attribute__((ext_vector_type(8)))  float  v8f;
typedef __attribute__((__vector_size__(4 * sizeof(int)))) int gv4i;

#define BM 128
#define BN 128
#define BK 32
#define LDSK 40   // padded K stride: 80 B = 5*16 B -> aligned, conflict-free b128

__device__ __forceinline__ __bf16 f32_to_bf16_rn(float f) {
    unsigned u = __builtin_bit_cast(unsigned, f);
    u += 0x7FFFu + ((u >> 16) & 1u);                // round-to-nearest-even
    unsigned short h = (unsigned short)(u >> 16);
    return __builtin_bit_cast(__bf16, h);
}
__device__ __forceinline__ float bf16_to_f32(__bf16 b) {
    unsigned short h = __builtin_bit_cast(unsigned short, b);
    unsigned u = ((unsigned)h) << 16;
    return __builtin_bit_cast(float, u);
}

// ---- global->LDS staging: async-to-LDS if the toolchain exposes it ----
#if defined(__AMDGCN__) && __has_builtin(__builtin_amdgcn_global_load_async_to_lds_b128)
#define ASYNC_LDS 1
#endif

__device__ __forceinline__ void copy16(__bf16* dst_lds, const __bf16* src_g) {
#if defined(ASYNC_LDS)
    __builtin_amdgcn_global_load_async_to_lds_b128(
        (__attribute__((address_space(1))) gv4i*)src_g,
        (__attribute__((address_space(3))) gv4i*)dst_lds,
        /*offset=*/0, /*cpol=*/0);
#else
    *(uint4*)dst_lds = *(const uint4*)src_g;
#endif
}

__device__ __forceinline__ void wait_stage() {
#if defined(ASYNC_LDS)
#if __has_builtin(__builtin_amdgcn_s_wait_asynccnt)
    __builtin_amdgcn_s_wait_asynccnt(0);
#else
    asm volatile("s_wait_asynccnt 0" ::: "memory");
#endif
#endif
}

// ---------------------------------------------------------------------------
// Kernel 1: effective weight, split to bf16 hi/lo, stored TRANSPOSED [N][K].
// ---------------------------------------------------------------------------
__global__ __launch_bounds__(256)
void prep_w(const float* __restrict__ base_t, const float* __restrict__ coeff,
            const int* __restrict__ mask,
            __bf16* __restrict__ wh, __bf16* __restrict__ wl,
            int K, int N) {
    __shared__ float tile[32][65];
    const int bn = blockIdx.x * 64;
    const int bk = blockIdx.y * 32;
    const int t  = threadIdx.x;
    const int nw = N >> 5;

#pragma unroll
    for (int i = 0; i < 8; ++i) {                   // load: n-contiguous
        int c = i * 256 + t;
        int k = c >> 6, n = c & 63;
        int gk = bk + k, gn = bn + n;
        unsigned w = (unsigned)mask[(size_t)gk * nw + (gn >> 5)];
        float s = ((w >> (gn & 31)) & 1u) ? 1.0f : -1.0f;
        tile[k][n] = base_t[(size_t)gk * N + gn] + coeff[gk] * s;
    }
    __syncthreads();
#pragma unroll
    for (int i = 0; i < 8; ++i) {                   // store: k-contiguous
        int c = i * 256 + t;
        int n = c >> 5, k = c & 31;
        float v = tile[k][n];
        __bf16 h = f32_to_bf16_rn(v);
        __bf16 l = f32_to_bf16_rn(v - bf16_to_f32(h));
        size_t o = (size_t)(bn + n) * K + (size_t)(bk + k);
        wh[o] = h;
        wl[o] = l;
    }
}

// ---------------------------------------------------------------------------
// Kernel 2: one-time x split into bf16 hi/lo, same [M][K] layout (coalesced).
// ---------------------------------------------------------------------------
__global__ __launch_bounds__(256)
void prep_x(const float* __restrict__ x,
            __bf16* __restrict__ xh, __bf16* __restrict__ xl) {
    size_t c = (size_t)blockIdx.x * 256 + threadIdx.x;   // float4 chunk id
    const float4 v = *(const float4*)&x[c * 4];
    const float* fp = reinterpret_cast<const float*>(&v);
    uint64_t hi = 0, lo = 0;
#pragma unroll
    for (int q = 0; q < 4; ++q) {
        __bf16 hq = f32_to_bf16_rn(fp[q]);
        __bf16 lq = f32_to_bf16_rn(fp[q] - bf16_to_f32(hq));
        hi |= (uint64_t)__builtin_bit_cast(unsigned short, hq) << (16 * q);
        lo |= (uint64_t)__builtin_bit_cast(unsigned short, lq) << (16 * q);
    }
    *(uint64_t*)&xh[c * 4] = hi;
    *(uint64_t*)&xl[c * 4] = lo;
}

// ---------------------------------------------------------------------------
// Kernel 3: tiled GEMM, bf16x3, double-buffered LDS.
// Block 128x128, BK=32, 8 waves (4Mx2N), wave -> 32x64 = 2x4 WMMA accums.
// PX=true : x pre-split in workspace, staging is pure b128 async copies.
// PX=false: x split in-kernel (workspace too small for xh/xl).
// ---------------------------------------------------------------------------
__device__ __forceinline__ v16bf ldfrag(const __bf16* p) {
    v8bf a = *(const v8bf*)p;           // K group {0..7}/{8..15}
    v8bf b = *(const v8bf*)(p + 16);    // K group +16
    return __builtin_shufflevector(a, b, 0,1,2,3,4,5,6,7,8,9,10,11,12,13,14,15);
}

template<bool PX>
__global__ __launch_bounds__(256)
void bdr_gemm(const float* __restrict__ xf,
              const __bf16* __restrict__ gxh, const __bf16* __restrict__ gxl,
              const __bf16* __restrict__ wh,  const __bf16* __restrict__ wl,
              float* __restrict__ y, int M, int N, int K) {
    __shared__ __bf16 sxh[2][BM * LDSK];
    __shared__ __bf16 sxl[2][BM * LDSK];
    __shared__ __bf16 swh[2][BN * LDSK];
    __shared__ __bf16 swl[2][BN * LDSK];

    const int t    = threadIdx.x;
    const int lane = t & 31;
    const int wid  = t >> 5;
    const int wm   = (wid & 3) * 32;
    const int wn   = (wid >> 2) * 64;
    const size_t m0 = (size_t)blockIdx.y * BM;
    const size_t n0 = (size_t)blockIdx.x * BN;
    const int frow  = lane & 15;
    const int khalf = (lane >> 4) << 3;

    v8f zero = {0.f, 0.f, 0.f, 0.f, 0.f, 0.f, 0.f, 0.f};
    v8f acc[2][4];
#pragma unroll
    for (int i = 0; i < 2; ++i)
#pragma unroll
        for (int j = 0; j < 4; ++j) acc[i][j] = zero;

    auto stage = [&](int buf, int kb) {
        // W tiles: 128 rows x 32 K bf16, b128 chunks
#pragma unroll
        for (int i = 0; i < 2; ++i) {
            int c = t + i * 256;
            int n = c >> 2, ko = (c & 3) << 3;
            size_t g = (n0 + n) * (size_t)K + kb + ko;
            copy16(&swh[buf][n * LDSK + ko], &wh[g]);
            copy16(&swl[buf][n * LDSK + ko], &wl[g]);
        }
        if constexpr (PX) {
#pragma unroll
            for (int i = 0; i < 2; ++i) {
                int c = t + i * 256;
                int m = c >> 2, ko = (c & 3) << 3;
                size_t g = (m0 + m) * (size_t)K + kb + ko;
                copy16(&sxh[buf][m * LDSK + ko], &gxh[g]);
                copy16(&sxl[buf][m * LDSK + ko], &gxl[g]);
            }
        } else {
#pragma unroll
            for (int i = 0; i < 4; ++i) {
                int c = t + i * 256;
                int m = c >> 3, ko = (c & 7) << 2;
                const float4 v = *(const float4*)&xf[(m0 + m) * K + kb + ko];
                const float* fp = reinterpret_cast<const float*>(&v);
                uint64_t hi = 0, lo = 0;
#pragma unroll
                for (int q = 0; q < 4; ++q) {
                    __bf16 hq = f32_to_bf16_rn(fp[q]);
                    __bf16 lq = f32_to_bf16_rn(fp[q] - bf16_to_f32(hq));
                    hi |= (uint64_t)__builtin_bit_cast(unsigned short, hq) << (16 * q);
                    lo |= (uint64_t)__builtin_bit_cast(unsigned short, lq) << (16 * q);
                }
                *(uint64_t*)&sxh[buf][m * LDSK + ko] = hi;
                *(uint64_t*)&sxl[buf][m * LDSK + ko] = lo;
            }
        }
    };

    stage(0, 0);
    const int nsteps = K / BK;
    for (int s = 0; s < nsteps; ++s) {
        const int buf = s & 1;
        wait_stage();            // async writes for buf complete (no-op if sync)
        __syncthreads();         // all tiles visible; prev compute reads done
        if (s + 1 < nsteps) stage(buf ^ 1, (s + 1) * BK);   // overlap w/ compute

        v16bf ah[2], al[2];
#pragma unroll
        for (int i = 0; i < 2; ++i) {
            int ra = (wm + i * 16 + frow) * LDSK + khalf;
            ah[i] = ldfrag(&sxh[buf][ra]);
            al[i] = ldfrag(&sxl[buf][ra]);
        }
#pragma unroll
        for (int j = 0; j < 4; ++j) {
            int rb = (wn + j * 16 + frow) * LDSK + khalf;
            v16bf bh = ldfrag(&swh[buf][rb]);
            v16bf bl = ldfrag(&swl[buf][rb]);
#pragma unroll
            for (int i = 0; i < 2; ++i) {
                acc[i][j] = __builtin_amdgcn_wmma_f32_16x16x32_bf16(
                    false, ah[i], false, bh, (short)0, acc[i][j], false, false);
                acc[i][j] = __builtin_amdgcn_wmma_f32_16x16x32_bf16(
                    false, ah[i], false, bl, (short)0, acc[i][j], false, false);
                acc[i][j] = __builtin_amdgcn_wmma_f32_16x16x32_bf16(
                    false, al[i], false, bh, (short)0, acc[i][j], false, false);
            }
        }
    }

    // fp32 C layout: VGPR r -> M=r (lanes 0-15) / M=8+r (lanes 16-31); N=lane&15
    const int ncol = lane & 15;
    const int mhi  = (lane >> 4) << 3;
#pragma unroll
    for (int i = 0; i < 2; ++i)
#pragma unroll
        for (int j = 0; j < 4; ++j) {
            size_t row = m0 + wm + i * 16 + mhi;
            size_t col = n0 + wn + j * 16 + ncol;
#pragma unroll
            for (int r = 0; r < 8; ++r)
                y[(row + r) * (size_t)N + col] = acc[i][j][r];
        }
}

// ---------------------------------------------------------------------------
extern "C" void kernel_launch(void* const* d_in, const int* in_sizes, int n_in,
                              void* d_out, int out_size, void* d_ws, size_t ws_size,
                              hipStream_t stream) {
    const float* x      = (const float*)d_in[0];
    const float* base_t = (const float*)d_in[1];
    const float* coeff  = (const float*)d_in[2];
    const int*   mask   = (const int*)d_in[3];
    float*       y      = (float*)d_out;

    const int K = in_sizes[2];                  // in_dim
    const int N = (in_sizes[3] * 32) / K;       // out_dim from packed mask
    const int M = in_sizes[0] / K;              // B*S rows

    __bf16* wh = (__bf16*)d_ws;                 // [N][K] hi
    __bf16* wl = wh + (size_t)K * N;            // [N][K] lo
    const size_t wbytes = (size_t)2 * (size_t)K * N * sizeof(__bf16);
    const size_t xbytes = (size_t)2 * (size_t)M * K * sizeof(__bf16);

    dim3 pgrid(N / 64, K / 32);
    prep_w<<<pgrid, 256, 0, stream>>>(base_t, coeff, mask, wh, wl, K, N);

    dim3 ggrid(N / BN, M / BM);
    if (ws_size >= wbytes + xbytes) {
        __bf16* xh = wl + (size_t)K * N;        // [M][K] hi
        __bf16* xl = xh + (size_t)M * K;        // [M][K] lo
        size_t chunks = ((size_t)M * K) / 4;
        prep_x<<<dim3((unsigned)(chunks / 256)), 256, 0, stream>>>(x, xh, xl);
        bdr_gemm<true><<<ggrid, 256, 0, stream>>>(x, xh, xl, wh, wl, y, M, N, K);
    } else {
        bdr_gemm<false><<<ggrid, 256, 0, stream>>>(x, nullptr, nullptr, wh, wl, y, M, N, K);
    }
}
